// mk1_Encoder_35579509080813
// MI455X (gfx1250) — compile-verified, matrix operand order checked by reading, not run on hardware
//
#include <hip/hip_runtime.h>
#include <hip/hip_bf16.h>
#include <math.h>

// ---------------------------------------------------------------------------
// Types for CDNA5 WMMA (wave32, 16x16x32 bf16 -> f32 accumulate)
// ---------------------------------------------------------------------------
typedef __attribute__((ext_vector_type(16))) __bf16 v16bf;
typedef __attribute__((ext_vector_type(8)))  float  v8f;

union MkU16 { unsigned int u[8]; v16bf v; };
union MkF8  { float f[8]; v8f v; };

#define GF_GELU 1
#define GF_ACC  2
#define GF_WB16 4
#define GF_WF32 8

#if __has_builtin(__builtin_amdgcn_global_load_async_to_lds_b128)
#define MK1_ASYNC 1
#else
#define MK1_ASYNC 0
#endif

__device__ __forceinline__ unsigned short mk1_f2bf(float f) {
  unsigned u = __float_as_uint(f);
  u += 0x7FFFu + ((u >> 16) & 1u);          // round-to-nearest-even
  return (unsigned short)(u >> 16);
}
__device__ __forceinline__ float mk1_gelu(float v) {
  return 0.5f * v * (1.0f + erff(v * 0.70710678118654752440f));
}

#if MK1_ASYNC
typedef int mk1_v4i __attribute__((vector_size(16)));
__device__ __forceinline__ void mk1_async_b128(const uint4* g, uint4* l) {
  __builtin_amdgcn_global_load_async_to_lds_b128(
      (__attribute__((address_space(1))) mk1_v4i*)(mk1_v4i*)g,
      (__attribute__((address_space(3))) mk1_v4i*)(mk1_v4i*)l, 0, 0);
}
__device__ __forceinline__ void mk1_wait_async() {
#if __has_builtin(__builtin_amdgcn_s_wait_asynccnt)
  __builtin_amdgcn_s_wait_asynccnt(0);
#else
  asm volatile("s_wait_asynccnt 0x0" ::: "memory");
#endif
}
#endif

// ---------------------------------------------------------------------------
// Tiled bf16 WMMA GEMM:  C[M,F] (+)= act( A[M,K]_bf16 @ Wt[F,K]_bf16^T + bias )
// Block: 256 threads = 8 waves; block tile = 256 rows; each wave owns TWO
// 16-row sub-tiles so every staged B fragment feeds two WMMAs. K chunk = 64,
// F chunk = NT*16 (NT=4 when F%64==0, NT=2 for the F=32 head).
// Tiles are staged with GLOBAL_LOAD_ASYNC_TO_LDS_B128 (ASYNCcnt) when the
// toolchain exposes it; otherwise uint4 VGPR staging.
// ---------------------------------------------------------------------------
template<int NT, int FLAGS>
__global__ __launch_bounds__(256)
void mk1_gemm(const unsigned short* __restrict__ A,
              const unsigned short* __restrict__ Wt,
              const float* __restrict__ bias,
              float* __restrict__ Cf,
              unsigned short* __restrict__ Cb,
              int M, int K, int F)
{
  __shared__ unsigned int sA[256 * 32];   // 256 rows x 64 bf16 (as 32 uints)
  __shared__ unsigned int sW[64 * 32];    // 64 cols  x 64 bf16
  const uint4* Au4 = (const uint4*)A;
  const uint4* Wu4 = (const uint4*)Wt;
  uint4* sA4 = (uint4*)sA;
  uint4* sW4 = (uint4*)sW;
  const int K4   = K >> 3;                // uint4 per row
  const int m0   = blockIdx.x * 256;
  const int wave = threadIdx.x >> 5;
  const int lane = threadIdx.x & 31;
  const int r16  = lane & 15;
  const int hh   = lane >> 4;
  const int h4   = hh * 4;

  for (int f0 = 0; f0 < F; f0 += NT * 16) {
    MkF8 acc[2][NT];
#pragma unroll
    for (int s = 0; s < 2; ++s)
#pragma unroll
      for (int ft = 0; ft < NT; ++ft) {
        if (FLAGS & GF_ACC) {
#pragma unroll
          for (int r = 0; r < 8; ++r) {
            int m = m0 + wave * 32 + s * 16 + hh * 8 + r;
            int n = f0 + ft * 16 + r16;
            acc[s][ft].f[r] = (m < M) ? Cf[(size_t)m * F + n] : 0.0f;
          }
        } else {
          float bv = bias ? bias[f0 + ft * 16 + r16] : 0.0f;
#pragma unroll
          for (int r = 0; r < 8; ++r) acc[s][ft].f[r] = bv;
        }
      }

    for (int k0 = 0; k0 < K; k0 += 64) {
#if MK1_ASYNC
#pragma unroll
      for (int it = 0; it < 8; ++it) {             // async stage A tile (256 x 64)
        int idx = threadIdx.x + it * 256;          // 2048 uint4
        int row = idx >> 3, k4 = idx & 7;
        int gr = m0 + row; if (gr > M - 1) gr = M - 1;
        mk1_async_b128(Au4 + (size_t)gr * K4 + (k0 >> 3) + k4, sA4 + idx);
      }
#pragma unroll
      for (int it = 0; it < 2; ++it) {             // async stage W tile (64 x 64)
        int idx = threadIdx.x + it * 256;          // 512 uint4
        int fr = idx >> 3, k4 = idx & 7;
        int gf = f0 + fr; if (gf > F - 1) gf = F - 1;   // padded rows never consumed
        mk1_async_b128(Wu4 + (size_t)gf * K4 + (k0 >> 3) + k4, sW4 + idx);
      }
      mk1_wait_async();
#else
#pragma unroll
      for (int it = 0; it < 8; ++it) {             // stage A tile (256 x 64) via B128
        int idx = threadIdx.x + it * 256;          // 2048 uint4
        int row = idx >> 3, k4 = idx & 7;
        int gr = m0 + row; if (gr > M - 1) gr = M - 1;
        sA4[idx] = Au4[(size_t)gr * K4 + (k0 >> 3) + k4];
      }
#pragma unroll
      for (int it = 0; it < 2; ++it) {             // stage W tile (64 x 64) via B128
        int idx = threadIdx.x + it * 256;          // 512 uint4
        int fr = idx >> 3, k4 = idx & 7;
        int gf = f0 + fr; if (gf > F - 1) gf = F - 1;
        sW4[idx] = Wu4[(size_t)gf * K4 + (k0 >> 3) + k4];
      }
      if (k0 + 64 < K) {                           // prefetch next A chunk
        int pr = m0 + (threadIdx.x >> 1); if (pr > M - 1) pr = M - 1;
        __builtin_prefetch(&Au4[(size_t)pr * K4 + ((k0 + 64) >> 3)], 0, 0);
      }
#endif
      __syncthreads();

#pragma unroll
      for (int kt = 0; kt < 2; ++kt) {
        MkU16 af[2];
#pragma unroll
        for (int s = 0; s < 2; ++s) {
          const unsigned int* ar = &sA[(wave * 32 + s * 16 + r16) * 32 + kt * 16];
#pragma unroll
          for (int v = 0; v < 4; ++v) { af[s].u[v] = ar[h4 + v]; af[s].u[4 + v] = ar[8 + h4 + v]; }
        }
#pragma unroll
        for (int ft = 0; ft < NT; ++ft) {
          MkU16 bfr;
          const unsigned int* br = &sW[(ft * 16 + r16) * 32 + kt * 16];
#pragma unroll
          for (int v = 0; v < 4; ++v) { bfr.u[v] = br[h4 + v]; bfr.u[4 + v] = br[8 + h4 + v]; }
#pragma unroll
          for (int s = 0; s < 2; ++s)
            acc[s][ft].v = __builtin_amdgcn_wmma_f32_16x16x32_bf16(
                false, af[s].v, false, bfr.v, (short)0, acc[s][ft].v, false, false);
        }
      }
      __syncthreads();
    }

#pragma unroll
    for (int s = 0; s < 2; ++s)
#pragma unroll
      for (int ft = 0; ft < NT; ++ft) {
#pragma unroll
        for (int r = 0; r < 8; ++r) {
          int m = m0 + wave * 32 + s * 16 + hh * 8 + r;
          if (m < M) {
            int n = f0 + ft * 16 + r16;
            float v = acc[s][ft].f[r];
            if (FLAGS & GF_GELU) v = mk1_gelu(v);
            if (FLAGS & GF_WF32) Cf[(size_t)m * F + n] = v;
            if (FLAGS & GF_WB16) Cb[(size_t)m * F + n] = mk1_f2bf(v);
          }
        }
      }
  }
}

static void launch_gemm(const unsigned short* A, const unsigned short* Wt, const float* bias,
                        float* Cf, unsigned short* Cb, int M, int K, int F, int flags,
                        hipStream_t stream) {
  dim3 g((M + 255) / 256), b(256);
  if (F % 64 == 0) {
    switch (flags) {
      case GF_GELU | GF_WB16:
        mk1_gemm<4, GF_GELU | GF_WB16><<<g, b, 0, stream>>>(A, Wt, bias, Cf, Cb, M, K, F); break;
      case GF_GELU | GF_WF32:
        mk1_gemm<4, GF_GELU | GF_WF32><<<g, b, 0, stream>>>(A, Wt, bias, Cf, Cb, M, K, F); break;
      case GF_WF32:
        mk1_gemm<4, GF_WF32><<<g, b, 0, stream>>>(A, Wt, bias, Cf, Cb, M, K, F); break;
      default: // GF_ACC | GF_WF32
        mk1_gemm<4, GF_ACC | GF_WF32><<<g, b, 0, stream>>>(A, Wt, bias, Cf, Cb, M, K, F); break;
    }
  } else {  // F == 32 head
    mk1_gemm<2, GF_GELU | GF_WF32><<<g, b, 0, stream>>>(A, Wt, bias, Cf, Cb, M, K, F);
  }
}

// ---------------------------------------------------------------------------
// Small utility / elementwise kernels
// ---------------------------------------------------------------------------
__global__ void mk1_zero(float* p, long long n) {
  for (long long i = blockIdx.x * 256LL + threadIdx.x; i < n; i += (long long)gridDim.x * 256)
    p[i] = 0.0f;
}

// Wt[f*Kpad+k] = bf16(W[k*F+f]); zero-pad k>=K
__global__ void mk1_prep_wt(const float* __restrict__ W, unsigned short* __restrict__ Wt,
                            int K, int F, int Kpad) {
  long long t = blockIdx.x * 256LL + threadIdx.x;
  if (t >= (long long)F * Kpad) return;
  int f = (int)(t / Kpad), k = (int)(t % Kpad);
  Wt[t] = (k < K) ? mk1_f2bf(W[(size_t)k * F + f]) : (unsigned short)0;
}

// Wt[f*256+k] = bf16(w0[k*256+f]+w1[..]+w2[..])
__global__ void mk1_prep_wrsum(const float* w0, const float* w1, const float* w2,
                               unsigned short* Wt) {
  int t = blockIdx.x * 256 + threadIdx.x;
  if (t >= 256 * 256) return;
  int f = t >> 8, k = t & 255;
  size_t i = (size_t)k * 256 + f;
  Wt[t] = mk1_f2bf(w0[i] + w1[i] + w2[i]);
}

__global__ void mk1_sum3(const float* a, const float* b, const float* c, float* o) {
  int t = threadIdx.x;
  if (t < 256) o[t] = a[t] + b[t] + c[t];
}

// per-column sum & sum-of-squares (one block per column)
__global__ __launch_bounds__(256)
void mk1_colstats(const float* __restrict__ x, int M, int C,
                  float* __restrict__ sums, float* __restrict__ sumsq) {
  int c = blockIdx.x;
  float s = 0.f, s2 = 0.f;
  for (int r = threadIdx.x; r < M; r += 256) {
    float v = x[(size_t)r * C + c];
    s += v; s2 += v * v;
  }
  __shared__ float bs[256], bq[256];
  bs[threadIdx.x] = s; bq[threadIdx.x] = s2; __syncthreads();
  for (int st = 128; st > 0; st >>= 1) {
    if (threadIdx.x < st) { bs[threadIdx.x] += bs[threadIdx.x + st]; bq[threadIdx.x] += bq[threadIdx.x + st]; }
    __syncthreads();
  }
  if (threadIdx.x == 0) { sums[c] = bs[0]; sumsq[c] = bq[0]; }
}

// batchnorm -> bf16
__global__ void mk1_bn_apply(const float* __restrict__ x, const float* __restrict__ sums,
                             const float* __restrict__ sumsq, const float* __restrict__ gamma,
                             const float* __restrict__ beta, unsigned short* __restrict__ out,
                             int M) {
  long long t = blockIdx.x * 256LL + threadIdx.x;
  if (t >= (long long)M * 64) return;
  int c = (int)(t & 63);
  float mu = sums[c] / M;
  float var = sumsq[c] / M - mu * mu;
  float y = gamma[c] * (x[t] - mu) * rsqrtf(var + 1e-5f) + beta[c];
  out[t] = mk1_f2bf(y);
}

// DynamicTanh with optional f32/bf16 outputs
__global__ void mk1_dyt(const float* __restrict__ in, const float* __restrict__ aP,
                        const float* __restrict__ w, const float* __restrict__ b,
                        float* __restrict__ outF, unsigned short* __restrict__ outB,
                        int C, long long count) {
  long long t = blockIdx.x * 256LL + threadIdx.x;
  if (t >= count) return;
  int c = (int)(t % C);
  float y = w[c] * tanhf(aP[0] * in[t]) + b[c];
  if (outF) outF[t] = y;
  if (outB) outB[t] = mk1_f2bf(y);
}

__global__ void mk1_geludiv3(const float* __restrict__ in, float* __restrict__ out, long long count) {
  long long t = blockIdx.x * 256LL + threadIdx.x;
  if (t >= count) return;
  out[t] = mk1_gelu(in[t] * 0.33333333333333333f);
}

// graphnorm + save DyT'd bf16 JK slice + bf16 copy for next layer
__global__ void mk1_gnapply(float* __restrict__ x, const float* __restrict__ sums,
                            const float* __restrict__ sumsq,
                            const float* __restrict__ nb, const float* __restrict__ nms,
                            const float* __restrict__ nw,
                            const float* __restrict__ d1a, const float* __restrict__ d1b,
                            const float* __restrict__ d1w,
                            unsigned short* __restrict__ xb, unsigned short* __restrict__ jkb,
                            int Loff, int M) {
  long long t = blockIdx.x * 256LL + threadIdx.x;
  if (t >= (long long)M * 256) return;
  int row = (int)(t >> 8), c = (int)(t & 255);
  float mu = sums[c] / M;
  float ex2 = sumsq[c] / M;
  float ms = nms[c];
  float var = ex2 - 2.0f * ms * mu * mu + ms * ms * mu * mu;
  float y = nw[c] * (x[t] - ms * mu) * rsqrtf(var + 1e-5f) + nb[c];
  x[t] = y;
  xb[t] = mk1_f2bf(y);
  int jc = Loff + c;
  jkb[(size_t)row * 768 + jc] = mk1_f2bf(d1w[jc] * tanhf(d1a[0] * y) + d1b[jc]);
}

// degree count & inverse
__global__ void mk1_deg(const int* __restrict__ dst, float* __restrict__ deg, int E) {
  int e = blockIdx.x * 256 + threadIdx.x;
  if (e < E) atomicAdd(&deg[dst[e]], 1.0f);
}
__global__ void mk1_invdeg(const float* __restrict__ deg, float* __restrict__ inv, int N) {
  int i = blockIdx.x * 256 + threadIdx.x;
  if (i < N) inv[i] = 1.0f / fmaxf(deg[i], 1.0f);
}

// scatter-add x[src] rows into agg[dst] rows (256-wide, 4 floats per thread)
__global__ void mk1_scatter(const int* __restrict__ src, const int* __restrict__ dst,
                            const float* __restrict__ x, float* __restrict__ agg, int E) {
  long long t = blockIdx.x * 256LL + threadIdx.x;
  if (t >= (long long)E * 64) return;
  int e = (int)(t >> 6);
  int c4 = (int)(t & 63) * 4;
  int s = src[e], d = dst[e];
  const float4 v = *(const float4*)&x[(size_t)s * 256 + c4];
  float* o = &agg[(size_t)d * 256 + c4];
  atomicAdd(o + 0, v.x); atomicAdd(o + 1, v.y);
  atomicAdd(o + 2, v.z); atomicAdd(o + 3, v.w);
}

__global__ void mk1_scaleb(const float* __restrict__ agg, const float* __restrict__ inv,
                           unsigned short* __restrict__ out, int M) {
  long long t = blockIdx.x * 256LL + threadIdx.x;
  if (t >= (long long)M * 256) return;
  out[t] = mk1_f2bf(agg[t] * inv[t >> 8]);
}

// DyT(enc) ++ x_AA ++ zero-pad  -> bf16 [N,320]
__global__ void mk1_lincat(const float* __restrict__ enc, const float* __restrict__ xaa,
                           const float* __restrict__ a2, const float* __restrict__ b2,
                           const float* __restrict__ w2, unsigned short* __restrict__ cat,
                           int M) {
  long long t = blockIdx.x * 256LL + threadIdx.x;
  if (t >= (long long)M * 320) return;
  int row = (int)(t / 320), c = (int)(t % 320);
  unsigned short o = 0;
  if (c < 256) {
    float v = enc[(size_t)row * 256 + c];
    o = mk1_f2bf(w2[c] * tanhf(a2[0] * v) + b2[c]);
  } else if (c < 276) {
    o = mk1_f2bf(xaa[(size_t)row * 20 + (c - 256)]);
  }
  cat[t] = o;
}

// vector quantize: argmin over 128 codes, emit q rows and sum of best sqdist
__global__ __launch_bounds__(256)
void mk1_vq(const float* __restrict__ z, const float* __restrict__ cb,
            float* __restrict__ q, float* __restrict__ lossacc, int M) {
  __shared__ float scb[128 * 32];
  for (int i = threadIdx.x; i < 4096; i += 256) scb[i] = cb[i];
  __syncthreads();
  int row = blockIdx.x * 256 + threadIdx.x;
  float local = 0.0f;
  if (row < M) {
    float zr[32];
#pragma unroll
    for (int k = 0; k < 32; ++k) zr[k] = z[(size_t)row * 32 + k];
    float bestd = 3.4e38f; int bi = 0;
    for (int c = 0; c < 128; ++c) {
      float d = 0.0f;
#pragma unroll
      for (int k = 0; k < 32; ++k) { float t = scb[c * 32 + k] - zr[k]; d += t * t; }
      if (d < bestd) { bestd = d; bi = c; }
    }
#pragma unroll
    for (int k = 0; k < 32; ++k) q[(size_t)row * 32 + k] = scb[bi * 32 + k];
    local = bestd;
  }
  __shared__ float red[256];
  red[threadIdx.x] = local; __syncthreads();
  for (int st = 128; st > 0; st >>= 1) {
    if (threadIdx.x < st) red[threadIdx.x] += red[threadIdx.x + st];
    __syncthreads();
  }
  if (threadIdx.x == 0) atomicAdd(lossacc, red[0]);
}

__global__ void mk1_fin(float* out, const float* lossacc, int M) {
  if (threadIdx.x == 0 && blockIdx.x == 0)
    out[0] = lossacc[0] * (1.25f / ((float)M * 32.0f));
}

// ---------------------------------------------------------------------------
// Host: parameter offsets follow JAX pytree (sorted-dict) flattening order.
// ---------------------------------------------------------------------------
extern "C" void kernel_launch(void* const* d_in, const int* in_sizes, int n_in,
                              void* d_out, int out_size, void* d_ws, size_t ws_size,
                              hipStream_t stream) {
  const int N = in_sizes[0] / 64;
  const float* x_res = (const float*)d_in[0];
  const float* x_AA  = (const float*)d_in[1];
  const int* ei[3] = { (const int*)d_in[2], (const int*)d_in[3], (const int*)d_in[4] };
  const int  Ec[3] = { in_sizes[2] / 2, in_sizes[3] / 2, in_sizes[4] / 2 };
  const float* P = (const float*)d_in[5];
  float* outF = (float*)d_out;

  // --- flat parameter offsets (sorted pytree) ---
  const long long O_BN_BETA = 0, O_BN_GAMMA = 64, O_CODEBOOK = 128;
  const long long O_CONVS = 4224;                       // per type: bl(256), wl(65536), wr(65536)
  const long long O_FF_B1 = 1186176, O_FF_B2 = 1186688;
  const long long O_FF_DA = 1186944, O_FF_DB = 1186945, O_FF_DW = 1187201;
  const long long O_FF_W1 = 1187457, O_FF_W2 = 1220225;
  const long long O_L_B1 = 1351297, O_L_B2 = 1351553;
  const long long O_L_D1A = 1351809, O_L_D1B = 1351810, O_L_D1W = 1352578;
  const long long O_L_D2A = 1353346, O_L_D2B = 1353347, O_L_D2W = 1353603;
  const long long O_L_W1 = 1353859, O_L_W2 = 1550467;
  const long long O_NORMS = 1616003;                    // per layer: b, ms, w (256 each)
  const long long O_O_B1 = 1618307, O_O_B2 = 1618563, O_O_B3 = 1618691;
  const long long O_O_DA = 1618723, O_O_DB = 1618724, O_O_DW = 1618756;
  const long long O_O_W1 = 1618788, O_O_W2 = 1689444, O_O_W3 = 1722212;
  auto cBL = [&](int L, int t) { return P + O_CONVS + (long long)L * 393984 + (long long)t * 131328; };

  // --- workspace arena ---
  char* ws = (char*)d_ws;
  size_t off = 0;
  auto alloc = [&](size_t bytes) -> char* {
    char* p = ws + off;
    off += (bytes + 255) & ~(size_t)255;
    return p;
  };
  unsigned short* wt_ff1 = (unsigned short*)alloc((size_t)512 * 64 * 2);
  unsigned short* wt_ff2 = (unsigned short*)alloc((size_t)256 * 512 * 2);
  unsigned short* wt_wr[3]; for (int L = 0; L < 3; ++L) wt_wr[L] = (unsigned short*)alloc((size_t)256 * 256 * 2);
  unsigned short* wt_wl[3][3];
  for (int L = 0; L < 3; ++L) for (int t = 0; t < 3; ++t) wt_wl[L][t] = (unsigned short*)alloc((size_t)256 * 256 * 2);
  float* bsum[3]; for (int L = 0; L < 3; ++L) bsum[L] = (float*)alloc(256 * 4);
  unsigned short* wt_l1 = (unsigned short*)alloc((size_t)256 * 768 * 2);
  unsigned short* wt_l2 = (unsigned short*)alloc((size_t)256 * 256 * 2);
  unsigned short* wt_o1 = (unsigned short*)alloc((size_t)256 * 320 * 2);
  unsigned short* wt_o2 = (unsigned short*)alloc((size_t)128 * 256 * 2);
  unsigned short* wt_o3 = (unsigned short*)alloc((size_t)32 * 128 * 2);
  float* stats = (float*)alloc(2 * 512 * 4);            // [sums(512), sumsq(512)]
  float* invdeg = (float*)alloc((size_t)3 * N * 4);
  float* lossacc = (float*)alloc(256);
  unsigned short* bnX = (unsigned short*)alloc((size_t)N * 64 * 2);
  unsigned short* h1  = (unsigned short*)alloc((size_t)N * 512 * 2);   // alias: e1b, catb
  float* xf  = (float*)alloc((size_t)N * 256 * 4);                     // alias: o3f
  unsigned short* xb = (unsigned short*)alloc((size_t)N * 256 * 2);    // alias: o2b
  float* accb = (float*)alloc((size_t)N * 256 * 4);                    // ff2 out, conv acc, z
  float* agg  = (float*)alloc((size_t)N * 256 * 4);                    // deg tmp, enc2f
  unsigned short* aggb = (unsigned short*)alloc((size_t)N * 256 * 2);  // alias: o1b
  unsigned short* jkb  = (unsigned short*)alloc((size_t)N * 768 * 2);
  unsigned short* e1b = h1;  unsigned short* catb = h1;
  float* enc2f = agg;        unsigned short* o1b = aggb;
  unsigned short* o2b = xb;  float* o3f = xf;  float* zf = accb;

  auto ge = [&](long long n) { return dim3((unsigned)((n + 255) / 256)); };
  auto gemm = [&](const unsigned short* A, const unsigned short* Wt, const float* bias,
                  float* Cf, unsigned short* Cb, int K, int F, int flags) {
    launch_gemm(A, Wt, bias, Cf, Cb, N, K, F, flags, stream);
  };
  auto zero = [&](float* p, long long n) { mk1_zero<<<dim3(2048), dim3(256), 0, stream>>>(p, n); };

  // --- weight prep (bf16, transposed) ---
  mk1_prep_wt<<<ge((long long)512 * 64), 256, 0, stream>>>(P + O_FF_W1, wt_ff1, 64, 512, 64);
  mk1_prep_wt<<<ge((long long)256 * 512), 256, 0, stream>>>(P + O_FF_W2, wt_ff2, 512, 256, 512);
  for (int L = 0; L < 3; ++L) {
    mk1_prep_wrsum<<<ge(256 * 256), 256, 0, stream>>>(cBL(L, 0) + 256 + 65536, cBL(L, 1) + 256 + 65536,
                                                      cBL(L, 2) + 256 + 65536, wt_wr[L]);
    for (int t = 0; t < 3; ++t)
      mk1_prep_wt<<<ge(256 * 256), 256, 0, stream>>>(cBL(L, t) + 256, wt_wl[L][t], 256, 256, 256);
    mk1_sum3<<<1, 256, 0, stream>>>(cBL(L, 0), cBL(L, 1), cBL(L, 2), bsum[L]);
  }
  mk1_prep_wt<<<ge((long long)256 * 768), 256, 0, stream>>>(P + O_L_W1, wt_l1, 768, 256, 768);
  mk1_prep_wt<<<ge(256 * 256), 256, 0, stream>>>(P + O_L_W2, wt_l2, 256, 256, 256);
  mk1_prep_wt<<<ge((long long)256 * 320), 256, 0, stream>>>(P + O_O_W1, wt_o1, 276, 256, 320);
  mk1_prep_wt<<<ge(128 * 256), 256, 0, stream>>>(P + O_O_W2, wt_o2, 256, 128, 256);
  mk1_prep_wt<<<ge(32 * 128), 256, 0, stream>>>(P + O_O_W3, wt_o3, 128, 32, 128);

  zero(lossacc, 1);

  // --- inverse degrees per edge type (agg[0..N) as temp) ---
  for (int t = 0; t < 3; ++t) {
    zero(agg, N);
    mk1_deg<<<ge(Ec[t]), 256, 0, stream>>>(ei[t] + Ec[t], agg, Ec[t]);
    mk1_invdeg<<<ge(N), 256, 0, stream>>>(agg, invdeg + (size_t)t * N, N);
  }

  // --- batchnorm + input MLP ---
  mk1_colstats<<<64, 256, 0, stream>>>(x_res, N, 64, stats, stats + 512);
  mk1_bn_apply<<<ge((long long)N * 64), 256, 0, stream>>>(x_res, stats, stats + 512,
                                                          P + O_BN_GAMMA, P + O_BN_BETA, bnX, N);
  gemm(bnX, wt_ff1, P + O_FF_B1, nullptr, h1, 64, 512, GF_GELU | GF_WB16);
  gemm(h1, wt_ff2, P + O_FF_B2, accb, nullptr, 512, 256, GF_GELU | GF_WF32);
  mk1_dyt<<<ge((long long)N * 256), 256, 0, stream>>>(accb, P + O_FF_DA, P + O_FF_DW, P + O_FF_DB,
                                                      xf, xb, 256, (long long)N * 256);

  // --- 3 hetero-SAGE layers ---
  for (int L = 0; L < 3; ++L) {
    gemm(xb, wt_wr[L], bsum[L], accb, nullptr, 256, 256, GF_WF32);       // x @ sum(Wr) + sum(bl)
    for (int t = 0; t < 3; ++t) {
      zero(agg, (long long)N * 256);
      mk1_scatter<<<ge((long long)Ec[t] * 64), 256, 0, stream>>>(ei[t], ei[t] + Ec[t], xf, agg, Ec[t]);
      mk1_scaleb<<<ge((long long)N * 256), 256, 0, stream>>>(agg, invdeg + (size_t)t * N, aggb, N);
      gemm(aggb, wt_wl[L][t], nullptr, accb, nullptr, 256, 256, GF_ACC | GF_WF32);
    }
    mk1_geludiv3<<<ge((long long)N * 256), 256, 0, stream>>>(accb, xf, (long long)N * 256);
    mk1_colstats<<<256, 256, 0, stream>>>(xf, N, 256, stats, stats + 512);
    mk1_gnapply<<<ge((long long)N * 256), 256, 0, stream>>>(
        xf, stats, stats + 512,
        P + O_NORMS + (long long)L * 768, P + O_NORMS + (long long)L * 768 + 256,
        P + O_NORMS + (long long)L * 768 + 512,
        P + O_L_D1A, P + O_L_D1B, P + O_L_D1W,
        xb, jkb, L * 256, N);
  }

  // --- JK MLP ---
  gemm(jkb, wt_l1, P + O_L_B1, nullptr, e1b, 768, 256, GF_GELU | GF_WB16);
  gemm(e1b, wt_l2, P + O_L_B2, enc2f, nullptr, 256, 256, GF_GELU | GF_WF32);
  mk1_lincat<<<ge((long long)N * 320), 256, 0, stream>>>(enc2f, x_AA, P + O_L_D2A, P + O_L_D2B,
                                                         P + O_L_D2W, catb, N);

  // --- output MLP ---
  gemm(catb, wt_o1, P + O_O_B1, nullptr, o1b, 320, 256, GF_GELU | GF_WB16);
  gemm(o1b, wt_o2, P + O_O_B2, nullptr, o2b, 256, 128, GF_GELU | GF_WB16);
  gemm(o2b, wt_o3, P + O_O_B3, o3f, nullptr, 128, 32, GF_GELU | GF_WF32);
  mk1_dyt<<<ge((long long)N * 32), 256, 0, stream>>>(o3f, P + O_O_DA, P + O_O_DW, P + O_O_DB,
                                                     zf, nullptr, 32, (long long)N * 32);

  // --- vector quantize + loss ---
  mk1_vq<<<ge(N), 256, 0, stream>>>(zf, P + O_CODEBOOK, outF, lossacc, N);
  mk1_fin<<<1, 1, 0, stream>>>(outF + (size_t)N * 32, lossacc, N);
}
